// MultiHeadAttention_75402445848963
// MI455X (gfx1250) — matrix operand order repaired and checked by reading, not compiled, not run on hardware
//
#include <hip/hip_runtime.h>

#define NJ      18
#define DIMQ    512
#define NHEAD   8
#define DDEPTH  64
#define NBATCH  4096
#define MROWS   (NBATCH * NJ)   /* 73728 */
#define KSTRIDE 520             /* padded LDS row stride (bf16 elems), 16B aligned */

typedef __attribute__((ext_vector_type(16))) __bf16 v16bf;
typedef __attribute__((ext_vector_type(8)))  __bf16 v8bf;
typedef __attribute__((ext_vector_type(8)))  float  v8f;
typedef __attribute__((ext_vector_type(4)))  float  v4f;

// Async global->LDS path (gfx1250 GLOBAL_LOAD_ASYNC_TO_LDS_B128), guarded so a
// missing builtin degrades to the plain copy path instead of failing compile.
#if __has_builtin(__builtin_amdgcn_global_load_async_to_lds_b128) && \
    __has_builtin(__builtin_amdgcn_s_wait_asynccnt)
#define USE_ASYNC_LDS 1
typedef int v4i_vs __attribute__((vector_size(16)));
typedef __attribute__((address_space(1))) v4i_vs* gas_ptr;
typedef __attribute__((address_space(3))) v4i_vs* las_ptr;
#endif

// ---------------------------------------------------------------------------
// WMMA fragment loaders (CDNA5 ISA 7.12.2 VGPR layouts, 16x16x32 bf16)
// ---------------------------------------------------------------------------

// A fragment (16x32) from fp32 row-major [*, 512], converted to bf16.
// lanes 0-15:  M=lane,     K in {0..7, 16..23}
// lanes 16-31: M=lane-16,  K in {8..15, 24..31}
__device__ __forceinline__ v16bf load_a_f32(const float* __restrict__ A,
                                            int row, int k0, int lane) {
  const int m  = lane & 15;
  const int kb = (lane >> 4) << 3;
  const float* p = A + (size_t)(row + m) * DIMQ + k0 + kb;
  v4f x0 = *(const v4f*)(p);
  v4f x1 = *(const v4f*)(p + 4);
  v4f y0 = *(const v4f*)(p + 16);
  v4f y1 = *(const v4f*)(p + 20);
  v16bf a;
#pragma unroll
  for (int i = 0; i < 4; ++i) {
    a[i]      = (__bf16)x0[i];
    a[4 + i]  = (__bf16)x1[i];
    a[8 + i]  = (__bf16)y0[i];
    a[12 + i] = (__bf16)y1[i];
  }
  return a;
}

// A fragment from bf16 row-major [*, 512].
__device__ __forceinline__ v16bf load_a_bf16(const __bf16* __restrict__ A,
                                             int row, int k0, int lane) {
  const int m  = lane & 15;
  const int kb = (lane >> 4) << 3;
  const __bf16* p = A + (size_t)(row + m) * DIMQ + k0 + kb;
  v8bf lo = *(const v8bf*)(p);
  v8bf hi = *(const v8bf*)(p + 16);
  v16bf a;
#pragma unroll
  for (int i = 0; i < 8; ++i) { a[i] = lo[i]; a[8 + i] = hi[i]; }
  return a;
}

// B fragment (32x16) from bf16 W^T stored [N][K] row-major.
// lanes 0-15: N=lane, K in {0..7, 16..23} ; lanes 16-31: N=lane-16, K in {8..15, 24..31}
__device__ __forceinline__ v16bf load_b(const __bf16* __restrict__ Wt,
                                        int col, int k0, int lane) {
  const int n  = lane & 15;
  const int kb = (lane >> 4) << 3;
  const __bf16* p = Wt + (size_t)(col + n) * DIMQ + k0 + kb;
  v8bf lo = *(const v8bf*)(p);
  v8bf hi = *(const v8bf*)(p + 16);
  v16bf b;
#pragma unroll
  for (int i = 0; i < 8; ++i) { b[i] = lo[i]; b[8 + i] = hi[i]; }
  return b;
}

// ---------------------------------------------------------------------------
// Weight transpose + fp32 -> bf16 : Wt[w][n][k] = W_w[k][n]
// ---------------------------------------------------------------------------
__global__ __launch_bounds__(256) void prep_weights(
    const float* __restrict__ Wq, const float* __restrict__ Wk,
    const float* __restrict__ Wv, const float* __restrict__ Wo,
    __bf16* __restrict__ Wt) {
  int idx = blockIdx.x * 256 + threadIdx.x;   // 4 * 512 * 512 total
  int w = idx >> 18;
  int r = idx & 262143;
  int n = r >> 9;
  int k = r & 511;
  const float* W = (w == 0) ? Wq : (w == 1) ? Wk : (w == 2) ? Wv : Wo;
  Wt[idx] = (__bf16)W[k * DIMQ + n];
}

// ---------------------------------------------------------------------------
// GEMM: C_bf16[73728,512] = A_f32[73728,512] @ W (via bf16 WMMA, f32 accum)
// block = 256 threads = 8 waves; block tile 128x64; wave tile 32x32
// ---------------------------------------------------------------------------
__global__ __launch_bounds__(256) void gemm_proj(
    const float* __restrict__ A, const __bf16* __restrict__ Wt,
    __bf16* __restrict__ C) {
  const int lane = threadIdx.x & 31;
  const int wave = threadIdx.x >> 5;
  const int bM = (blockIdx.x >> 3) * 128;
  const int bN = (blockIdx.x & 7) * 64;
  const int row = bM + (wave >> 1) * 32;
  const int col = bN + (wave & 1) * 32;

  v8f c00 = {}, c01 = {}, c10 = {}, c11 = {};
#pragma unroll 4
  for (int k0 = 0; k0 < DIMQ; k0 += 32) {
    // prefetch the K-step after next of the streamed A rows (-> global_prefetch_b8, WGP scope)
    if (k0 + 64 < DIMQ) {
      __builtin_prefetch(A + (size_t)(row + (lane & 15)) * DIMQ + k0 + 64, 0, 3);
      __builtin_prefetch(A + (size_t)(row + 16 + (lane & 15)) * DIMQ + k0 + 64, 0, 3);
    }
    v16bf a0 = load_a_f32(A, row,      k0, lane);
    v16bf a1 = load_a_f32(A, row + 16, k0, lane);
    v16bf b0 = load_b(Wt, col,      k0, lane);
    v16bf b1 = load_b(Wt, col + 16, k0, lane);
    c00 = __builtin_amdgcn_wmma_f32_16x16x32_bf16(false, a0, false, b0, (short)0, c00, false, false);
    c01 = __builtin_amdgcn_wmma_f32_16x16x32_bf16(false, a0, false, b1, (short)0, c01, false, false);
    c10 = __builtin_amdgcn_wmma_f32_16x16x32_bf16(false, a1, false, b0, (short)0, c10, false, false);
    c11 = __builtin_amdgcn_wmma_f32_16x16x32_bf16(false, a1, false, b1, (short)0, c11, false, false);
  }
  const int n  = lane & 15;
  const int mh = (lane >> 4) << 3;
#pragma unroll
  for (int r = 0; r < 8; ++r) {
    C[(size_t)(row + mh + r) * DIMQ + col + n]           = (__bf16)c00[r];
    C[(size_t)(row + mh + r) * DIMQ + col + 16 + n]      = (__bf16)c01[r];
    C[(size_t)(row + 16 + mh + r) * DIMQ + col + n]      = (__bf16)c10[r];
    C[(size_t)(row + 16 + mh + r) * DIMQ + col + 16 + n] = (__bf16)c11[r];
  }
}

// ---------------------------------------------------------------------------
// Final GEMM: out_f32 = attout_bf16 @ Wo + bo
// ---------------------------------------------------------------------------
__global__ __launch_bounds__(256) void gemm_out(
    const __bf16* __restrict__ A, const __bf16* __restrict__ Wt,
    const float* __restrict__ bo, float* __restrict__ Out) {
  const int lane = threadIdx.x & 31;
  const int wave = threadIdx.x >> 5;
  const int bM = (blockIdx.x >> 3) * 128;
  const int bN = (blockIdx.x & 7) * 64;
  const int row = bM + (wave >> 1) * 32;
  const int col = bN + (wave & 1) * 32;

  v8f c00 = {}, c01 = {}, c10 = {}, c11 = {};
#pragma unroll 4
  for (int k0 = 0; k0 < DIMQ; k0 += 32) {
    if (k0 + 64 < DIMQ) {
      __builtin_prefetch(A + (size_t)(row + (lane & 15)) * DIMQ + k0 + 64, 0, 3);
      __builtin_prefetch(A + (size_t)(row + 16 + (lane & 15)) * DIMQ + k0 + 64, 0, 3);
    }
    v16bf a0 = load_a_bf16(A, row,      k0, lane);
    v16bf a1 = load_a_bf16(A, row + 16, k0, lane);
    v16bf b0 = load_b(Wt, col,      k0, lane);
    v16bf b1 = load_b(Wt, col + 16, k0, lane);
    c00 = __builtin_amdgcn_wmma_f32_16x16x32_bf16(false, a0, false, b0, (short)0, c00, false, false);
    c01 = __builtin_amdgcn_wmma_f32_16x16x32_bf16(false, a0, false, b1, (short)0, c01, false, false);
    c10 = __builtin_amdgcn_wmma_f32_16x16x32_bf16(false, a1, false, b0, (short)0, c10, false, false);
    c11 = __builtin_amdgcn_wmma_f32_16x16x32_bf16(false, a1, false, b1, (short)0, c11, false, false);
  }
  const int n  = lane & 15;
  const int mh = (lane >> 4) << 3;
  const float bia0 = bo[col + n];
  const float bia1 = bo[col + 16 + n];
#pragma unroll
  for (int r = 0; r < 8; ++r) {
    Out[(size_t)(row + mh + r) * DIMQ + col + n]           = c00[r] + bia0;
    Out[(size_t)(row + mh + r) * DIMQ + col + 16 + n]      = c01[r] + bia1;
    Out[(size_t)(row + 16 + mh + r) * DIMQ + col + n]      = c10[r] + bia0;
    Out[(size_t)(row + 16 + mh + r) * DIMQ + col + 16 + n] = c11[r] + bia1;
  }
}

// ---------------------------------------------------------------------------
// Fused attention: one block (8 wave32s) per batch element; wave h == head h.
//   rel_bias[b,i,j] = t[i][map[i,j]],  t[i][m] = dot(sum_h K[b,:,i,:], rel_k[m])
//   out[i,d]       += sum_j att[i,j] * (V[j,d] + rel_v[map[i,j]][d])
// ---------------------------------------------------------------------------
__global__ __launch_bounds__(256) void attn_fused(
    const __bf16* __restrict__ Qp, const __bf16* __restrict__ Kp,
    const __bf16* __restrict__ Vp,
    const float* __restrict__ rel_k, const float* __restrict__ rel_v,
    const int* __restrict__ jmap,
    float* __restrict__ att_out, __bf16* __restrict__ attout) {
  __shared__ __bf16 sQ[NJ * KSTRIDE];
  __shared__ __bf16 sK[NJ * KSTRIDE];
  __shared__ __bf16 sV[NJ * KSTRIDE];
  __shared__ float  sRk[4 * DDEPTH];
  __shared__ float  sRv[4 * DDEPTH];
  __shared__ float  sT[NJ * 4];
  __shared__ int    sMap[NJ * NJ];

  const int b    = blockIdx.x;
  const int tid  = threadIdx.x;
  const size_t base = (size_t)b * NJ * DIMQ;

  // stage Q/K/V rows (bf16, padded stride) into LDS
#if defined(USE_ASYNC_LDS)
  for (int idx = tid; idx < NJ * (DIMQ / 8); idx += 256) {
    int i = idx >> 6, c = idx & 63;
    int dst = i * KSTRIDE + c * 8;
    __builtin_amdgcn_global_load_async_to_lds_b128(
        (gas_ptr)(__bf16*)(Qp + base + (size_t)idx * 8), (las_ptr)(sQ + dst), 0, 0);
    __builtin_amdgcn_global_load_async_to_lds_b128(
        (gas_ptr)(__bf16*)(Kp + base + (size_t)idx * 8), (las_ptr)(sK + dst), 0, 0);
    __builtin_amdgcn_global_load_async_to_lds_b128(
        (gas_ptr)(__bf16*)(Vp + base + (size_t)idx * 8), (las_ptr)(sV + dst), 0, 0);
  }
#else
  for (int idx = tid; idx < NJ * (DIMQ / 8); idx += 256) {
    int i = idx >> 6, c = idx & 63;
    *(v8bf*)(sQ + i * KSTRIDE + c * 8) = ((const v8bf*)(Qp + base))[idx];
    *(v8bf*)(sK + i * KSTRIDE + c * 8) = ((const v8bf*)(Kp + base))[idx];
    *(v8bf*)(sV + i * KSTRIDE + c * 8) = ((const v8bf*)(Vp + base))[idx];
  }
#endif
  if (tid < 4 * DDEPTH) { sRk[tid] = rel_k[tid]; sRv[tid] = rel_v[tid]; }
  for (int idx = tid; idx < NJ * NJ; idx += 256) sMap[idx] = jmap[idx];
#if defined(USE_ASYNC_LDS)
  __builtin_amdgcn_s_wait_asynccnt(0);
#endif
  __syncthreads();

  // t[i][m] = sum_d (sum_h K[i, h*64+d]) * rel_k[m][d]
  for (int idx = tid; idx < NJ * 4; idx += 256) {
    int i = idx >> 2, m = idx & 3;
    float s = 0.0f;
    for (int d = 0; d < DDEPTH; ++d) {
      float ks = 0.0f;
#pragma unroll
      for (int hh = 0; hh < NHEAD; ++hh) ks += (float)sK[i * KSTRIDE + hh * DDEPTH + d];
      s += ks * sRk[m * DDEPTH + d];
    }
    sT[idx] = s;
  }
  __syncthreads();

  const int h    = tid >> 5;     // head == wave
  const int lane = tid & 31;
  const float scale = 0.125f;    // 64^-0.5
  float attreg[NJ];

#pragma unroll
  for (int i = 0; i < NJ; ++i) {
    float logit = -3.0e38f;
    if (lane < NJ) {
      float s = 0.0f;
#pragma unroll
      for (int d = 0; d < DDEPTH; ++d)
        s += (float)sQ[i * KSTRIDE + h * DDEPTH + d] *
             (float)sK[lane * KSTRIDE + h * DDEPTH + d];
      logit = (s + sT[i * 4 + sMap[i * NJ + lane]]) * scale;
    }
    float mx = logit;
    for (int off = 16; off > 0; off >>= 1) mx = fmaxf(mx, __shfl_xor(mx, off, 32));
    float e = (lane < NJ) ? __expf(logit - mx) : 0.0f;
    float sm = e;
    for (int off = 16; off > 0; off >>= 1) sm += __shfl_xor(sm, off, 32);
    float a = e / sm;
    attreg[i] = a;
    if (lane < NJ)
      att_out[(((size_t)b * NHEAD + h) * NJ + i) * NJ + lane] = a;
  }

  // out[i, h*64+d] = sum_j att[i][j] * (V[j,d] + rel_v[map[i,j]][d])
#pragma unroll
  for (int i = 0; i < NJ; ++i) {
#pragma unroll
    for (int db = 0; db < 2; ++db) {
      const int d = db * 32 + lane;
      float acc = 0.0f;
#pragma unroll
      for (int j = 0; j < NJ; ++j) {
        float av = __shfl(attreg[i], j, 32);
        acc += av * ((float)sV[j * KSTRIDE + h * DDEPTH + d] +
                     sRv[sMap[i * NJ + j] * DDEPTH + d]);
      }
      attout[base + (size_t)i * DIMQ + h * DDEPTH + d] = (__bf16)acc;
    }
  }
}

// ---------------------------------------------------------------------------
extern "C" void kernel_launch(void* const* d_in, const int* in_sizes, int n_in,
                              void* d_out, int out_size, void* d_ws, size_t ws_size,
                              hipStream_t stream) {
  (void)in_sizes; (void)n_in; (void)out_size; (void)ws_size;
  const float* k_in = (const float*)d_in[0];
  const float* v_in = (const float*)d_in[1];
  const float* q_in = (const float*)d_in[2];
  const float* Wq   = (const float*)d_in[3];
  const float* Wk   = (const float*)d_in[4];
  const float* Wv   = (const float*)d_in[5];
  const float* Wo   = (const float*)d_in[6];
  const float* bo   = (const float*)d_in[7];
  const float* rk   = (const float*)d_in[8];
  const float* rv   = (const float*)d_in[9];
  const int*   jm   = (const int*)d_in[10];

  __bf16* Wt = (__bf16*)d_ws;                      // 4 * 512*512
  __bf16* Qp = Wt + (size_t)4 * DIMQ * DIMQ;
  __bf16* Kp = Qp + (size_t)MROWS * DIMQ;
  __bf16* Vp = Kp + (size_t)MROWS * DIMQ;
  __bf16* AO = Vp + (size_t)MROWS * DIMQ;

  float* outp = (float*)d_out;
  float* attp = outp + (size_t)MROWS * DIMQ;

  const int gemm_blocks = (MROWS / 128) * (DIMQ / 64);   // 4608

  prep_weights<<<4096, 256, 0, stream>>>(Wq, Wk, Wv, Wo, Wt);
  gemm_proj<<<gemm_blocks, 256, 0, stream>>>(q_in, Wt + 0 * (DIMQ * DIMQ), Qp);
  gemm_proj<<<gemm_blocks, 256, 0, stream>>>(k_in, Wt + 1 * (DIMQ * DIMQ), Kp);
  gemm_proj<<<gemm_blocks, 256, 0, stream>>>(v_in, Wt + 2 * (DIMQ * DIMQ), Vp);
  attn_fused<<<NBATCH, 256, 0, stream>>>(Qp, Kp, Vp, rk, rv, jm, attp, AO);
  gemm_out<<<gemm_blocks, 256, 0, stream>>>(AO, Wt + 3 * (DIMQ * DIMQ), bo, outp);
}
